// BlockDiagonalGRU_68728066671057
// MI455X (gfx1250) — matrix operand order repaired and checked by reading, not compiled, not run on hardware
//
#include <hip/hip_runtime.h>

typedef __attribute__((ext_vector_type(16))) __bf16 v16bf;
typedef __attribute__((ext_vector_type(8)))  __bf16 v8bf;
typedef __attribute__((ext_vector_type(8)))  float  v8f;

static constexpr int BATCH = 16384;
static constexpr int NB    = 8;
static constexpr int BLK   = 256;   // hidden block / in block
static constexpr int HID   = 2048;
static constexpr int GATES = 768;   // 3 * BLK

static constexpr int MT    = 64;    // batch rows per workgroup
static constexpr int NT    = 128;   // j-columns per workgroup (gate triples)
static constexpr int LDA   = 40;    // padded LDS row stride (bf16 elems, 80B => 16B aligned)

__device__ __forceinline__ v16bf cat16(v8bf lo, v8bf hi) {
    return __builtin_shufflevector(lo, hi, 0,1,2,3,4,5,6,7,8,9,10,11,12,13,14,15);
}

#if __has_builtin(__builtin_amdgcn_tanhf)
__device__ __forceinline__ float fast_tanh(float v) { return __builtin_amdgcn_tanhf(v); }
#else
__device__ __forceinline__ float fast_tanh(float v) {
    const float e2 = __expf(2.0f * v);
    return 1.0f - 2.0f * __builtin_amdgcn_rcpf(e2 + 1.0f);
}
#endif

__device__ __forceinline__ float fast_sigmoid(float v) {
    return __builtin_amdgcn_rcpf(1.0f + __expf(-v));
}

__global__ __launch_bounds__(256) void gru_block_wmma(
    const float* __restrict__ x,    const float* __restrict__ h,
    const float* __restrict__ w_ih, const float* __restrict__ w_hh,
    const float* __restrict__ b_ih, const float* __restrict__ b_hh,
    float* __restrict__ out)
{
    __shared__ __align__(16) __bf16 As[MT * LDA];        //  5.0 KB
    __shared__ __align__(16) __bf16 Bs[3 * NT * LDA];    // 30.0 KB

    const int t    = threadIdx.x;
    const int lane = t & 31;
    const int wave = t >> 5;

    const int m0 = blockIdx.x * MT;          // batch tile base
    const int n  = blockIdx.y >> 1;          // block index 0..7
    const int j0 = (blockIdx.y & 1) * NT;    // j-column tile base within block

    // staging indices: A tile = 64 rows x 32 K, one 8-float chunk per thread
    const int a_row = t >> 2;
    const int a_kc  = (t & 3) * 8;

    const float* const wih_n = w_ih + (size_t)n * GATES * BLK;
    const float* const whh_n = w_hh + (size_t)n * GATES * BLK;
    const size_t a_base = (size_t)(m0 + a_row) * HID + (size_t)n * BLK + a_kc;

    v8f acc[4][3] = {};   // [m-rep][gate], f32 16x16 accumulators

    const int jloc = wave * 16 + (lane & 15);   // local j column owned by this lane
    const int kb   = lane >> 4;                 // K-half select (A/B lane layout)

    // ---- software pipeline registers: next step's global data ----
    float4 fa0, fa1;
    float4 fb0[6], fb1[6];

    auto prefetch = [&](int it) {
        const int k0 = (it & 7) * 32;
        // uniform ternary keeps addrspace(1) => global_load_b128, not flat
        const float* ap = (it < 8 ? x : h) + a_base + k0;
        fa0 = ((const float4*)ap)[0];
        fa1 = ((const float4*)ap)[1];
        const float* wp = (it < 8 ? wih_n : whh_n);
        #pragma unroll
        for (int i = 0; i < 6; ++i) {
            const int c    = t + i * 256;
            const int row  = c >> 2;           // 0..383
            const int kc   = (c & 3) * 8;
            const int gate = row >> 7;         // 0..2
            const int jl   = row & 127;
            const float* p = wp + (unsigned)((gate * BLK + j0 + jl) * BLK + k0 + kc);
            fb0[i] = ((const float4*)p)[0];
            fb1[i] = ((const float4*)p)[1];
        }
    };

    auto stage = [&]() {
        __bf16* d = &As[a_row * LDA + a_kc];
        d[0]=(__bf16)fa0.x; d[1]=(__bf16)fa0.y; d[2]=(__bf16)fa0.z; d[3]=(__bf16)fa0.w;
        d[4]=(__bf16)fa1.x; d[5]=(__bf16)fa1.y; d[6]=(__bf16)fa1.z; d[7]=(__bf16)fa1.w;
        #pragma unroll
        for (int i = 0; i < 6; ++i) {
            const int c    = t + i * 256;
            const int row  = c >> 2;
            const int kc   = (c & 3) * 8;
            const int gate = row >> 7;
            const int jl   = row & 127;
            __bf16* db = &Bs[(gate * NT + jl) * LDA + kc];
            db[0]=(__bf16)fb0[i].x; db[1]=(__bf16)fb0[i].y;
            db[2]=(__bf16)fb0[i].z; db[3]=(__bf16)fb0[i].w;
            db[4]=(__bf16)fb1[i].x; db[5]=(__bf16)fb1[i].y;
            db[6]=(__bf16)fb1[i].z; db[7]=(__bf16)fb1[i].w;
        }
    };

    prefetch(0);
    for (int it = 0; it < 16; ++it) {
        __syncthreads();                    // previous step's fragment reads done
        stage();                            // registers -> bf16 LDS
        __syncthreads();                    // staging visible to all waves
        prefetch(it < 15 ? it + 1 : 15);    // issue next step's HBM loads now;
                                            // latency hidden under the WMMAs below
        // ---- B fragments (32x16 bf16, per-wave N columns) ----
        v16bf bfrag[3];
        #pragma unroll
        for (int g = 0; g < 3; ++g) {
            const __bf16* bp = &Bs[(g * NT + jloc) * LDA + kb * 16];
            bfrag[g] = cat16(*(const v8bf*)(bp), *(const v8bf*)(bp + 8));
        }
        // ---- A fragments (16x32 bf16) + 12 WMMAs ----
        #pragma unroll
        for (int r = 0; r < 4; ++r) {
            const __bf16* ap2 = &As[(r * 16 + (lane & 15)) * LDA + kb * 16];
            const v16bf afrag = cat16(*(const v8bf*)(ap2), *(const v8bf*)(ap2 + 8));
            #pragma unroll
            for (int g = 0; g < 3; ++g) {
                acc[r][g] = __builtin_amdgcn_wmma_f32_16x16x32_bf16(
                    false, afrag, false, bfrag[g], (short)0, acc[r][g], false, false);
            }
        }
    }

    // ---- epilogue: bias + GRU nonlinearity + blend with h ----
    const int jj = j0 + jloc;                    // column within block, 0..255
    const size_t bb = (size_t)n * GATES;
    const float br = b_ih[bb +         jj] + b_hh[bb +         jj];
    const float bu = b_ih[bb + BLK   + jj] + b_hh[bb + BLK   + jj];
    const float bc = b_ih[bb + 2*BLK + jj] + b_hh[bb + 2*BLK + jj];
    const int rowoff = (lane >> 4) * 8;          // C/D layout: M = vgpr + 8*(lane>=16)

    #pragma unroll
    for (int r = 0; r < 4; ++r) {
        #pragma unroll
        for (int v = 0; v < 8; ++v) {
            const int m = m0 + r * 16 + rowoff + v;
            const size_t idx = (size_t)m * HID + (size_t)n * BLK + jj;
            const float g0 = acc[r][0][v] + br;
            const float g1 = acc[r][1][v] + bu;
            const float g2 = acc[r][2][v] + bc;
            const float rr = fast_sigmoid(g0);
            const float uu = fast_sigmoid(g1);
            const float cc = fast_tanh(rr * g2);
            const float ho = h[idx];
            out[idx] = (1.0f - uu) * ho + uu * cc;
        }
    }
}

extern "C" void kernel_launch(void* const* d_in, const int* in_sizes, int n_in,
                              void* d_out, int out_size, void* d_ws, size_t ws_size,
                              hipStream_t stream) {
    const float* x    = (const float*)d_in[0];
    const float* h    = (const float*)d_in[1];
    const float* w_ih = (const float*)d_in[2];
    const float* w_hh = (const float*)d_in[3];
    const float* b_ih = (const float*)d_in[4];
    const float* b_hh = (const float*)d_in[5];
    float* out = (float*)d_out;
    (void)in_sizes; (void)n_in; (void)out_size; (void)d_ws; (void)ws_size;

    dim3 grid(BATCH / MT, NB * (BLK / NT));   // 256 x 16 workgroups
    gru_block_wmma<<<grid, 256, 0, stream>>>(x, h, w_ih, w_hh, b_ih, b_hh, out);
}
